// LSTMWithAttention_81741817577837
// MI455X (gfx1250) — matrix operand order repaired and checked by reading, not compiled, not run on hardware
//
#include <hip/hip_runtime.h>
#include <hip/hip_bf16.h>

#ifndef __HIP_MEMORY_SCOPE_AGENT
#define __HIP_MEMORY_SCOPE_AGENT 4
#endif

// ---------------------------------------------------------------------------
// Problem constants (from the reference)
// ---------------------------------------------------------------------------
#define BB   64
#define TT   512
#define IN_  64
#define HH   512
#define OUT_ 64
#define G4H  2048   // 4*H
#define NWG  16     // persistent grid: 16 workgroups x 8 waves
#define LDW  520    // LDS row stride (f16) for W_hh slices: 512+8 pad -> no bank conflicts
#define LDWI 72     // LDS row stride (f16) for W_ih0 slice: 64+8 pad  -> no bank conflicts

typedef __attribute__((ext_vector_type(16))) _Float16 v16h;
typedef __attribute__((ext_vector_type(8)))  float    v8f;

// ---------------------------------------------------------------------------
// WMMA fragment loaders (contiguous k order k = 16h + s, same for A and B; wave32)
//   A (16x32 f16, row-major source): lane L<16 -> row L, K = kb..kb+15
//     contiguous; lanes 16..31 -> same rows, K = kb+16..+31.
//   B (32x16 f16, column n contiguous in memory = row n of W): lane L<16 ->
//     col L, K = kb..kb+15 contiguous; lanes 16..31 -> col L-16, K = kb+16..+31.
// ---------------------------------------------------------------------------
__device__ __forceinline__ v16h frag_A(const _Float16* rowbase, int ld, int kb) {
    const int lane = threadIdx.x & 31;
    const _Float16* p = rowbase + (size_t)(lane & 15) * ld + kb + ((lane & 16) ? 16 : 0);
    union { v16h v; uint4 q[2]; } u;
    u.q[0] = *(const uint4*)(p);
    u.q[1] = *(const uint4*)(p + 8);
    return u.v;
}

__device__ __forceinline__ v16h frag_B(const _Float16* W, int rowbase, int K, int kb) {
    const int lane = threadIdx.x & 31;
    const _Float16* p = W + (size_t)(rowbase + (lane & 15)) * K + kb + ((lane & 16) ? 16 : 0);
    union { v16h v; uint4 q[2]; } u;
    u.q[0] = *(const uint4*)(p);
    u.q[1] = *(const uint4*)(p + 8);
    return u.v;
}

// B fragment from an LDS-staged weight slice with compile-time row stride
template <int STRIDE>
__device__ __forceinline__ v16h frag_B_lds(const _Float16* Ws, int rowbase, int kb) {
    const int lane = threadIdx.x & 31;
    const _Float16* p = Ws + (size_t)(rowbase + (lane & 15)) * STRIDE + kb + ((lane & 16) ? 16 : 0);
    union { v16h v; uint4 q[2]; } u;
    u.q[0] = *(const uint4*)(p);
    u.q[1] = *(const uint4*)(p + 8);
    return u.v;
}

__device__ __forceinline__ v8f wmma_f16f32(v16h a, v16h b, v8f c) {
    return __builtin_amdgcn_wmma_f32_16x16x32_f16(false, a, false, b, (short)0, c,
                                                  false, false);
}

__device__ __forceinline__ float sigmoidf_(float x) {
    x = fminf(fmaxf(x, -30.f), 30.f);
    return 1.f / (1.f + __expf(-x));
}
__device__ __forceinline__ float tanhf_(float x) {
    x = fminf(fmaxf(x, -15.f), 15.f);
    float e = __expf(-2.f * x);
    return (1.f - e) / (1.f + e);
}

// Device-wide barrier for the persistent kernel: monotonic counter, one
// arrival per block per phase; counter is zeroed by the host each call so
// graph replays are deterministic.
__device__ __forceinline__ void grid_sync(unsigned* ctr, unsigned phase) {
    __syncthreads();
    if (threadIdx.x == 0) {
        __threadfence();  // release my stores (h state) to device scope
        __hip_atomic_fetch_add(ctr, 1u, __ATOMIC_ACQ_REL, __HIP_MEMORY_SCOPE_AGENT);
        const unsigned target = phase * (unsigned)gridDim.x;
        while (__hip_atomic_load(ctr, __ATOMIC_ACQUIRE, __HIP_MEMORY_SCOPE_AGENT) < target) {
            __builtin_amdgcn_s_sleep(1);
        }
    }
    __syncthreads();
}

// ---------------------------------------------------------------------------
// Utility kernels
// ---------------------------------------------------------------------------
__global__ void cvt_f16_kernel(const float* __restrict__ in, _Float16* __restrict__ out, int n) {
    int i = blockIdx.x * blockDim.x + threadIdx.x;
    if (i < n) out[i] = (_Float16)in[i];
}

__global__ void bias_sum_kernel(const float* __restrict__ a, const float* __restrict__ b,
                                float* __restrict__ o, int n) {
    int i = blockIdx.x * blockDim.x + threadIdx.x;
    if (i < n) o[i] = a[i] + b[i];
}

__global__ void zero_u32_kernel(unsigned* __restrict__ p, int n) {
    int i = blockIdx.x * blockDim.x + threadIdx.x;
    if (i < n) p[i] = 0u;
}

// ---------------------------------------------------------------------------
// Persistent 2-layer LSTM recurrence. Grid: NWG(16) blocks x 256 threads.
// Each wave owns one (M-tile, 16-H-column) position and all 4 gate tiles;
// cell state c lives in registers for the entire sequence. Staged in LDS per
// WG (once): both layers' W_hh slices + the W_ih0 slice (all B operands that
// are small enough to be loop-invariant-hoisted and would otherwise spill).
// ---------------------------------------------------------------------------
__global__ __launch_bounds__(256) void lstm_persistent_kernel(
    const _Float16* __restrict__ x16,   // [B,T,IN] f16
    const _Float16* __restrict__ Wih0,  // [4H x IN]  f16
    const _Float16* __restrict__ Whh0,  // [4H x H]   f16
    const _Float16* __restrict__ Wih1,  // [4H x H]   f16
    const _Float16* __restrict__ Whh1,  // [4H x H]   f16
    const float*    __restrict__ bias0, // [4H]
    const float*    __restrict__ bias1, // [4H]
    _Float16* __restrict__ h0a, _Float16* __restrict__ h0b,  // [B,H] ping-pong
    _Float16* __restrict__ h1a, _Float16* __restrict__ h1b,  // [B,H] ping-pong
    _Float16* __restrict__ hs2,         // [B,T,H] layer-1 output sequence
    unsigned* __restrict__ barrier_ctr)
{
    extern __shared__ _Float16 smem[];  // 2*128*LDW + 128*LDWI f16 = 284,672 B
    _Float16* whh0_s = smem;
    _Float16* whh1_s = smem + 128 * LDW;
    _Float16* wih0_s = smem + 2 * 128 * LDW;

    const int wave = threadIdx.x >> 5;
    const int lane = threadIdx.x & 31;
    const int mtile = wave & 3;                  // 0..3  over B=64 rows
    const int ntl   = wave >> 2;                 // 0..1  col-tile within WG
    const int hbase0 = blockIdx.x * 32;          // WG's H-column base
    const int hbase  = hbase0 + ntl * 16;
    const int mbase  = mtile * 16;
    const int col    = lane & 15;
    const int rowoff = (lane & 16) ? 8 : 0;

    // ---- stage weight slices into LDS (one time) ----
    // local row r (0..127) <-> global row  gate(r>>5)*H + hbase0 + (r&31)
    for (int r = wave; r < 128; r += 8) {
        const int grow = (r >> 5) * HH + hbase0 + (r & 31);
        const uint4* s0 = (const uint4*)(Whh0 + (size_t)grow * HH);
        const uint4* s1 = (const uint4*)(Whh1 + (size_t)grow * HH);
        const uint4* s2 = (const uint4*)(Wih0 + (size_t)grow * IN_);
        uint4* d0 = (uint4*)(whh0_s + r * LDW);
        uint4* d1 = (uint4*)(whh1_s + r * LDW);
        uint4* d2 = (uint4*)(wih0_s + r * LDWI);
        for (int j = lane; j < 64; j += 32) { d0[j] = s0[j]; d1[j] = s1[j]; }
        if (lane < 8) d2[lane] = s2[lane];       // 8 x uint4 = 64 f16 per row
    }
    __syncthreads();

    // ---- per-wave loop-invariant state ----
    float b0[4], b1[4];
#pragma unroll
    for (int g = 0; g < 4; ++g) {
        b0[g] = bias0[g * HH + hbase + col];
        b1[g] = bias1[g * HH + hbase + col];
    }
    float c0r[8], c1r[8];
#pragma unroll
    for (int i = 0; i < 8; ++i) { c0r[i] = 0.f; c1r[i] = 0.f; }

    const _Float16* h0p = h0a; _Float16* h0n = h0b;
    const _Float16* h1p = h1a; _Float16* h1n = h1b;
    unsigned phase = 0;

    const int lrow = ntl * 16;   // this wave's base row inside the LDS slices

    for (int t = 0; t < TT; ++t) {
        // ================= layer 0 =================
        v8f acc[4] = {};
        {
            // input projection: x[:, t, :] · Wih0^T   (Kin = 64, B from LDS)
            const _Float16* Arow = x16 + (size_t)t * IN_ + (size_t)mbase * (TT * IN_);
            for (int kb = 0; kb < IN_; kb += 32) {
                v16h a = frag_A(Arow, TT * IN_, kb);
#pragma unroll
                for (int g = 0; g < 4; ++g) {
                    v16h b = frag_B_lds<LDWI>(wih0_s, g * 32 + lrow, kb);
                    acc[g] = wmma_f16f32(a, b, acc[g]);
                }
            }
            // recurrent projection: h0 · Whh0^T (B from LDS)
            const _Float16* Hrow = h0p + (size_t)mbase * HH;
            for (int kb = 0; kb < HH; kb += 32) {
                v16h a = frag_A(Hrow, HH, kb);
#pragma unroll
                for (int g = 0; g < 4; ++g) {
                    v16h b = frag_B_lds<LDW>(whh0_s, g * 32 + lrow, kb);
                    acc[g] = wmma_f16f32(a, b, acc[g]);
                }
            }
        }
#pragma unroll
        for (int i = 0; i < 8; ++i) {
            const int row = mbase + rowoff + i;
            float gi = sigmoidf_(acc[0][i] + b0[0]);
            float gf = sigmoidf_(acc[1][i] + b0[1]);
            float gg = tanhf_   (acc[2][i] + b0[2]);
            float go = sigmoidf_(acc[3][i] + b0[3]);
            c0r[i] = gf * c0r[i] + gi * gg;
            h0n[(size_t)row * HH + hbase + col] = (_Float16)(go * tanhf_(c0r[i]));
        }
        grid_sync(barrier_ctr, ++phase);

        // ================= layer 1 =================
        v8f acc1[4] = {};
        {
            // input projection: h0_new · Wih1^T  (Kin = 512, B from global/L2;
            // 64 distinct fragments per wave -> not hoistable, stays in-loop)
            const _Float16* Arow = h0n + (size_t)mbase * HH;
            for (int kb = 0; kb < HH; kb += 32) {
                v16h a = frag_A(Arow, HH, kb);
#pragma unroll
                for (int g = 0; g < 4; ++g) {
                    v16h b = frag_B(Wih1, g * HH + hbase, HH, kb);
                    acc1[g] = wmma_f16f32(a, b, acc1[g]);
                }
            }
            // recurrent projection: h1 · Whh1^T (B from LDS)
            const _Float16* Hrow = h1p + (size_t)mbase * HH;
            for (int kb = 0; kb < HH; kb += 32) {
                v16h a = frag_A(Hrow, HH, kb);
#pragma unroll
                for (int g = 0; g < 4; ++g) {
                    v16h b = frag_B_lds<LDW>(whh1_s, g * 32 + lrow, kb);
                    acc1[g] = wmma_f16f32(a, b, acc1[g]);
                }
            }
        }
#pragma unroll
        for (int i = 0; i < 8; ++i) {
            const int row = mbase + rowoff + i;
            float gi = sigmoidf_(acc1[0][i] + b1[0]);
            float gf = sigmoidf_(acc1[1][i] + b1[1]);
            float gg = tanhf_   (acc1[2][i] + b1[2]);
            float go = sigmoidf_(acc1[3][i] + b1[3]);
            c1r[i] = gf * c1r[i] + gi * gg;
            _Float16 h16 = (_Float16)(go * tanhf_(c1r[i]));
            h1n[(size_t)row * HH + hbase + col] = h16;
            hs2[((size_t)row * TT + t) * HH + hbase + col] = h16;
        }
        grid_sync(barrier_ctr, ++phase);

        // swap ping-pong buffers
        { const _Float16* tp = h0p; h0p = h0n; h0n = (_Float16*)tp; }
        { const _Float16* tp = h1p; h1p = h1n; h1n = (_Float16*)tp; }
    }
}

// ---------------------------------------------------------------------------
// Attention pooling: a[b,t] = (Watt[:H]+Watt[H:])·h2[b,t] + b_att
//                    pooled[b,:] = sum_t a[b,t] * h2[b,t,:]
// ---------------------------------------------------------------------------
__global__ __launch_bounds__(256) void attention_pool_kernel(
    const _Float16* __restrict__ hs,      // [B,T,H] f16
    const float* __restrict__ Watt,       // [2H]
    const float* __restrict__ batt_p,     // [1]
    _Float16* __restrict__ pooled16)      // [B,H] f16
{
    const int b = blockIdx.x;
    const int tid = threadIdx.x;
    const int lane = tid & 31, wave = tid >> 5;

    const float w0 = Watt[tid]       + Watt[HH + tid];
    const float w1 = Watt[tid + 256] + Watt[HH + tid + 256];
    const float batt = batt_p[0];
    const _Float16* base = hs + (size_t)b * TT * HH;

    float p0 = 0.f, p1 = 0.f;
    __shared__ float red[8];
    __shared__ float abc;

    for (int t = 0; t < TT; ++t) {
        float h0v = (float)base[(size_t)t * HH + tid];
        float h1v = (float)base[(size_t)t * HH + tid + 256];
        float part = w0 * h0v + w1 * h1v;
#pragma unroll
        for (int off = 16; off > 0; off >>= 1)
            part += __shfl_down(part, off, 32);
        if (lane == 0) red[wave] = part;
        __syncthreads();
        if (tid == 0) {
            float s = 0.f;
#pragma unroll
            for (int k = 0; k < 8; ++k) s += red[k];
            abc = s + batt;
        }
        __syncthreads();
        float a = abc;
        p0 += a * h0v;
        p1 += a * h1v;
        __syncthreads();
    }
    pooled16[(size_t)b * HH + tid]       = (_Float16)p0;
    pooled16[(size_t)b * HH + tid + 256] = (_Float16)p1;
}

// ---------------------------------------------------------------------------
// Final FC: out[64,64] = pooled[64,512] · Wfc^T + b_fc   (f16 WMMA, f32 accum)
// ---------------------------------------------------------------------------
__global__ __launch_bounds__(512) void fc_kernel(
    const _Float16* __restrict__ pooled16, // [B,H]
    const _Float16* __restrict__ Wfc16,    // [OUT,H] row-major
    const float* __restrict__ bfc,         // [OUT]
    float* __restrict__ out)               // [B,OUT]
{
    const int wave = threadIdx.x >> 5;
    const int lane = threadIdx.x & 31;
    const int mtile = wave >> 2, ntile = wave & 3;
    const int mbase = mtile * 16, nbase = ntile * 16;
    const int col = lane & 15;

    v8f acc = {};
    const _Float16* Arow = pooled16 + (size_t)mbase * HH;
    for (int kb = 0; kb < HH; kb += 32) {
        v16h a = frag_A(Arow, HH, kb);
        v16h b = frag_B(Wfc16, nbase, HH, kb);
        acc = wmma_f16f32(a, b, acc);
    }

    const float bv = bfc[nbase + col];
    const int rowoff = (lane & 16) ? 8 : 0;
#pragma unroll
    for (int i = 0; i < 8; ++i) {
        int row = mbase + rowoff + i;
        out[(size_t)row * OUT_ + nbase + col] = acc[i] + bv;
    }
}

// ---------------------------------------------------------------------------
// Host side
// ---------------------------------------------------------------------------
extern "C" void kernel_launch(void* const* d_in, const int* in_sizes, int n_in,
                              void* d_out, int out_size, void* d_ws, size_t ws_size,
                              hipStream_t stream) {
    (void)in_sizes; (void)n_in; (void)out_size; (void)ws_size;

    const float* x    = (const float*)d_in[0];
    const float* Wih0 = (const float*)d_in[1];
    const float* Whh0 = (const float*)d_in[2];
    const float* bih0 = (const float*)d_in[3];
    const float* bhh0 = (const float*)d_in[4];
    const float* Wih1 = (const float*)d_in[5];
    const float* Whh1 = (const float*)d_in[6];
    const float* bih1 = (const float*)d_in[7];
    const float* bhh1 = (const float*)d_in[8];
    const float* Watt = (const float*)d_in[9];
    const float* batt = (const float*)d_in[10];
    const float* Wfc  = (const float*)d_in[11];
    const float* bfc  = (const float*)d_in[12];
    float* out = (float*)d_out;

    // ---- workspace layout ----
    char* ws = (char*)d_ws;
    size_t off = 0;
    auto alloc = [&](size_t bytes) { void* p = ws + off; off += (bytes + 255) & ~(size_t)255; return p; };
    _Float16* x16     = (_Float16*)alloc((size_t)BB * TT * IN_ * 2);
    _Float16* Wih0_16 = (_Float16*)alloc((size_t)G4H * IN_ * 2);
    _Float16* Whh0_16 = (_Float16*)alloc((size_t)G4H * HH * 2);
    _Float16* Wih1_16 = (_Float16*)alloc((size_t)G4H * HH * 2);
    _Float16* Whh1_16 = (_Float16*)alloc((size_t)G4H * HH * 2);
    _Float16* Wfc16   = (_Float16*)alloc((size_t)OUT_ * HH * 2);
    float*    bias0   = (float*)alloc((size_t)G4H * 4);
    float*    bias1   = (float*)alloc((size_t)G4H * 4);
    // zero-init region: barrier counter + h0a + h1a (contiguous)
    unsigned* barrier_ctr = (unsigned*)alloc(256);
    _Float16* h0a     = (_Float16*)alloc((size_t)BB * HH * 2);
    _Float16* h1a     = (_Float16*)alloc((size_t)BB * HH * 2);
    // end zero-init region
    _Float16* h0b     = (_Float16*)alloc((size_t)BB * HH * 2);
    _Float16* h1b     = (_Float16*)alloc((size_t)BB * HH * 2);
    _Float16* pooled16= (_Float16*)alloc((size_t)BB * HH * 2);
    _Float16* hs2     = (_Float16*)alloc((size_t)BB * TT * HH * 2);

    auto cvt = [&](const float* in, _Float16* o, int n) {
        cvt_f16_kernel<<<(n + 255) / 256, 256, 0, stream>>>(in, o, n);
    };

    // ---- precision conversion + bias prep ----
    cvt(x,    x16,     BB * TT * IN_);
    cvt(Wih0, Wih0_16, G4H * IN_);
    cvt(Whh0, Whh0_16, G4H * HH);
    cvt(Wih1, Wih1_16, G4H * HH);
    cvt(Whh1, Whh1_16, G4H * HH);
    cvt(Wfc,  Wfc16,   OUT_ * HH);
    bias_sum_kernel<<<(G4H + 255) / 256, 256, 0, stream>>>(bih0, bhh0, bias0, G4H);
    bias_sum_kernel<<<(G4H + 255) / 256, 256, 0, stream>>>(bih1, bhh1, bias1, G4H);

    // zero barrier counter + h0a + h1a (contiguous)
    {
        int words = (256 + 2 * BB * HH * 2) / 4;
        zero_u32_kernel<<<(words + 255) / 256, 256, 0, stream>>>((unsigned*)barrier_ctr, words);
    }

    // ---- the whole 512-step, 2-layer recurrence in ONE persistent launch ----
    const size_t lds_bytes = (size_t)(2 * 128 * LDW + 128 * LDWI) * sizeof(_Float16); // 284,672 B
    lstm_persistent_kernel<<<NWG, 256, lds_bytes, stream>>>(
        x16, Wih0_16, Whh0_16, Wih1_16, Whh1_16, bias0, bias1,
        h0a, h0b, h1a, h1b, hs2, barrier_ctr);

    // ---- attention pooling + final FC ----
    attention_pool_kernel<<<BB, 256, 0, stream>>>(hs2, Watt, batt, pooled16);
    fc_kernel<<<1, 512, 0, stream>>>(pooled16, Wfc16, bfc, out);
}
